// GQAttention_49374944035439
// MI455X (gfx1250) — compile-verified, matrix operand order checked by reading, not run on hardware
//
#include <hip/hip_runtime.h>
#include <hip/hip_bf16.h>

// Problem constants (from reference)
#define B_    2
#define T_    2048
#define D_    1024
#define QH_   16
#define KVH_  4
#define HD_   64
#define G_    4
#define KVD_  512            // 2*KVH*HD
#define SCALE_ 0.125f        // 64^-0.5

// GEMM block tiling
#define BM 64
#define BN 128
#define BK 32

typedef __attribute__((ext_vector_type(16))) _Float16 v16h;
typedef __attribute__((ext_vector_type(8)))  _Float16 v8h;
typedef __attribute__((ext_vector_type(8)))  float    v8f;
typedef __attribute__((ext_vector_type(4)))  int      v4i;

#define WMMA_F16(a, b, c)                                                    \
  __builtin_amdgcn_wmma_f32_16x16x32_f16(false, (a), false, (b), (short)0,   \
                                         (c), false, false)

// ---------------------------------------------------------------------------
// CDNA5 async global->LDS copy (16 bytes per lane), tracked by ASYNCcnt.
// Builtin signature (from clang diagnostics): (v4i addrspace(1)*,
// v4i addrspace(3)*, imm offset, imm cpol).
// ---------------------------------------------------------------------------
#if __has_builtin(__builtin_amdgcn_global_load_async_to_lds_b128)
#define HAVE_ASYNC_LDS 1
#define ASYNC_CP16(gsrc, ldst)                                               \
  __builtin_amdgcn_global_load_async_to_lds_b128(                            \
      (__attribute__((address_space(1))) v4i*)(void*)(gsrc),                 \
      (__attribute__((address_space(3))) v4i*)(void*)(ldst), 0, 0)
#if __has_builtin(__builtin_amdgcn_s_wait_asynccnt)
#define ASYNC_WAIT() __builtin_amdgcn_s_wait_asynccnt(0)
#else
#define ASYNC_WAIT() asm volatile("s_wait_asynccnt 0" ::: "memory")
#endif
#else
#define HAVE_ASYNC_LDS 0
#define ASYNC_CP16(gsrc, ldst) (*(v8h*)(ldst) = *(const v8h*)(gsrc))
#define ASYNC_WAIT()
#endif

// ---------------------------------------------------------------------------
// Fragment loader for 16-bit A/B WMMA operands (CDNA5 ISA 7.12.2 layout).
// lanes 0-15 : elems 0-7 -> K = c+0..7,   elems 8-15 -> K = c+16..23
// lanes 16-31: elems 0-7 -> K = c+8..15,  elems 8-15 -> K = c+24..31
// Both halves are contiguous 16-byte chunks -> global_load_b128 / ds_load_b128.
// ---------------------------------------------------------------------------
static __device__ __forceinline__ v16h load_frag(const _Float16* base,
                                                 int row, int pitch, int c,
                                                 int lane) {
  const _Float16* p = base + (size_t)row * pitch + c + ((lane & 16) ? 8 : 0);
  v8h lo = *(const v8h*)(p);
  v8h hi = *(const v8h*)(p + 16);
  v16h r;
#pragma unroll
  for (int i = 0; i < 8; ++i) { r[i] = lo[i]; r[i + 8] = hi[i]; }
  return r;
}

static __device__ __forceinline__ float grp16_max(float v) {
  v = fmaxf(v, __shfl_xor(v, 1));
  v = fmaxf(v, __shfl_xor(v, 2));
  v = fmaxf(v, __shfl_xor(v, 4));
  v = fmaxf(v, __shfl_xor(v, 8));
  return v;
}
static __device__ __forceinline__ float grp16_sum(float v) {
  v += __shfl_xor(v, 1);
  v += __shfl_xor(v, 2);
  v += __shfl_xor(v, 4);
  v += __shfl_xor(v, 8);
  return v;
}

// ---------------------------------------------------------------------------
// Stage 0: fp32 -> fp16 conversion
// ---------------------------------------------------------------------------
__global__ __launch_bounds__(256) void f32_to_f16_kernel(
    const float* __restrict__ in, _Float16* __restrict__ out, int n) {
  int i = blockIdx.x * blockDim.x + threadIdx.x;
  if (i < n) out[i] = (_Float16)in[i];
}

// ---------------------------------------------------------------------------
// QKV epilogue: RoPE on the WMMA C fragment + scatter to attention layouts.
// cb (16-aligned column base) selects Q / K / V segment; uniform per wave.
// ---------------------------------------------------------------------------
static __device__ __forceinline__ void qkv_store_tile(
    v8f acc, int rb, int cb, int lane, const float* __restrict__ cosp,
    const float* __restrict__ sinp, _Float16* __restrict__ qh,
    _Float16* __restrict__ kh, _Float16* __restrict__ vt) {
  const int lr  = lane & 15;
  const int hi8 = (lane & 16) ? 8 : 0;
  if (cb < D_) {                         // ---- Q, apply RoPE ----
    const int h = cb >> 6;
    const int d = (cb & 63) + lr;
#pragma unroll
    for (int r = 0; r < 8; ++r) {
      const int m = rb + r + hi8;
      const int bt = m / T_, t = m % T_;
      float v   = acc[r];
      float prt = __shfl_xor(v, 1);      // RoPE partner (adjacent feature)
      float cs  = cosp[t * HD_ + d];
      float sn  = sinp[t * HD_ + d];
      float rot = (d & 1) ? prt : -prt;
      qh[(((size_t)bt * QH_ + h) * T_ + t) * HD_ + d] =
          (_Float16)(v * cs + rot * sn);
    }
  } else {
    const int f0  = cb - D_;
    const int kvh = f0 >> 7;
    const int d   = (f0 & 63) + lr;
    if (((f0 >> 6) & 1) == 0) {          // ---- K, apply RoPE ----
#pragma unroll
      for (int r = 0; r < 8; ++r) {
        const int m = rb + r + hi8;
        const int bt = m / T_, t = m % T_;
        float v   = acc[r];
        float prt = __shfl_xor(v, 1);
        float cs  = cosp[t * HD_ + d];
        float sn  = sinp[t * HD_ + d];
        float rot = (d & 1) ? prt : -prt;
        kh[(((size_t)bt * KVH_ + kvh) * T_ + t) * HD_ + d] =
            (_Float16)(v * cs + rot * sn);
      }
    } else {                             // ---- V, store transposed ----
#pragma unroll
      for (int r = 0; r < 8; ++r) {
        const int m = rb + r + hi8;
        const int bt = m / T_, t = m % T_;
        vt[(((size_t)bt * KVH_ + kvh) * HD_ + d) * T_ + t] = (_Float16)acc[r];
      }
    }
  }
}

// ---------------------------------------------------------------------------
// Stage 1: fused QKV projection + RoPE, block-tiled with async LDS staging.
// Block = 256 threads (8 waves) computing a 64(M) x 128(N) tile.
// Wave (wm,wn) computes a 2x2 grid of 16x16 WMMA accumulators.
// K loop: BK=32 slabs, double-buffered GLOBAL_LOAD_ASYNC_TO_LDS_B128.
// ---------------------------------------------------------------------------
__global__ __launch_bounds__(256) void qkv_rope_wmma(
    const _Float16* __restrict__ xh,   // (B*T, D)
    const _Float16* __restrict__ wqh,  // (D, D)
    const _Float16* __restrict__ wkvh, // (KVD, D)
    const float* __restrict__ cosp,    // (T, HD)
    const float* __restrict__ sinp,    // (T, HD)
    _Float16* __restrict__ qh,         // (B, QH, T, HD)
    _Float16* __restrict__ kh,         // (B, KVH, T, HD)
    _Float16* __restrict__ vt) {       // (B, KVH, HD, T)
  __shared__ _Float16 As[2][BM * BK];  // (m, k) row-major
  __shared__ _Float16 Bs[2][BN * BK];  // (n, k) row-major
  const int tid  = threadIdx.x;
  const int lane = tid & 31;
  const int wave = tid >> 5;
  const int lr   = lane & 15;
  const int m0   = blockIdx.x * BM;
  const int n0   = blockIdx.y * BN;
  const int wm   = (wave & 1) * 32;    // wave row offset in block tile
  const int wn   = (wave >> 1) * 32;   // wave col offset in block tile

  const _Float16* wbase = (n0 < D_) ? (wqh + (size_t)n0 * D_)
                                    : (wkvh + (size_t)(n0 - D_) * D_);
  // per-thread 16B copy lanes: A tile once, B tile twice
  const int cr = tid >> 2;             // row 0..63
  const int cc = (tid & 3) * 8;        // col 0,8,16,24 (halves)

  v8f a00 = {}, a01 = {}, a10 = {}, a11 = {};

  auto fill = [&](int buf, int k0) {
    ASYNC_CP16(xh + (size_t)(m0 + cr) * D_ + k0 + cc, &As[buf][cr * BK + cc]);
    ASYNC_CP16(wbase + (size_t)cr * D_ + k0 + cc, &Bs[buf][cr * BK + cc]);
    ASYNC_CP16(wbase + (size_t)(64 + cr) * D_ + k0 + cc,
               &Bs[buf][(64 + cr) * BK + cc]);
  };

  fill(0, 0);
  ASYNC_WAIT();
  __syncthreads();

  for (int k0 = 0; k0 < D_; k0 += BK) {
    const int buf = (k0 >> 5) & 1;
    if (k0 + BK < D_) fill(buf ^ 1, k0 + BK);    // prefetch next slab
    v16h fa0 = load_frag(As[buf], wm + lr,      BK, 0, lane);
    v16h fa1 = load_frag(As[buf], wm + 16 + lr, BK, 0, lane);
    v16h fb0 = load_frag(Bs[buf], wn + lr,      BK, 0, lane);
    v16h fb1 = load_frag(Bs[buf], wn + 16 + lr, BK, 0, lane);
    a00 = WMMA_F16(fa0, fb0, a00);
    a01 = WMMA_F16(fa0, fb1, a01);
    a10 = WMMA_F16(fa1, fb0, a10);
    a11 = WMMA_F16(fa1, fb1, a11);
    ASYNC_WAIT();
    __syncthreads();
  }

  qkv_store_tile(a00, m0 + wm,      n0 + wn,      lane, cosp, sinp, qh, kh, vt);
  qkv_store_tile(a01, m0 + wm,      n0 + wn + 16, lane, cosp, sinp, qh, kh, vt);
  qkv_store_tile(a10, m0 + wm + 16, n0 + wn,      lane, cosp, sinp, qh, kh, vt);
  qkv_store_tile(a11, m0 + wm + 16, n0 + wn + 16, lane, cosp, sinp, qh, kh, vt);
}

// ---------------------------------------------------------------------------
// Stage 2: flash attention. One wave per (b, head, 16-query tile).
// Streams keys 32 at a time: S(16x32)=Q·K^T via 4 WMMAs, online softmax,
// P goes through LDS (C-layout -> A-layout reshape), O(16x64)+=P·V via 4 WMMAs.
// ---------------------------------------------------------------------------
__global__ __launch_bounds__(32) void attn_wmma(
    const _Float16* __restrict__ qh, const _Float16* __restrict__ kh,
    const _Float16* __restrict__ vt, _Float16* __restrict__ ah) {
  __shared__ _Float16 pl[16 * 32];      // P tile staging (row-major 16x32)
  const int lane = threadIdx.x;
  const int lr   = lane & 15;
  const int hi8  = (lane & 16) ? 8 : 0;
  const int mt   = blockIdx.x;          // query tile (T/16)
  const int h    = blockIdx.y;          // q head
  const int bb   = blockIdx.z;          // batch
  const int kvh  = h >> 2;              // h / G

  const _Float16* qb = qh + (((size_t)bb * QH_ + h) * T_ + mt * 16) * HD_;
  const _Float16* kb = kh + ((size_t)bb * KVH_ + kvh) * (size_t)T_ * HD_;
  const _Float16* vb = vt + ((size_t)bb * KVH_ + kvh) * (size_t)HD_ * T_;

  v16h qa0 = load_frag(qb, lr, HD_, 0, lane);
  v16h qa1 = load_frag(qb, lr, HD_, 32, lane);

  v8f o0 = {}, o1 = {}, o2 = {}, o3 = {};
  float mrun[8], lrun[8];
#pragma unroll
  for (int r = 0; r < 8; ++r) { mrun[r] = -3.0e38f; lrun[r] = 0.0f; }

  for (int kt = 0; kt < T_; kt += 32) {
    v8f s0 = {}, s1 = {};
    {
      v16h ka = load_frag(kb + (size_t)kt * HD_, lr, HD_, 0, lane);
      v16h kc = load_frag(kb + (size_t)kt * HD_, lr, HD_, 32, lane);
      s0 = WMMA_F16(qa0, ka, s0);
      s0 = WMMA_F16(qa1, kc, s0);
      v16h kd = load_frag(kb + (size_t)(kt + 16) * HD_, lr, HD_, 0, lane);
      v16h ke = load_frag(kb + (size_t)(kt + 16) * HD_, lr, HD_, 32, lane);
      s1 = WMMA_F16(qa0, kd, s1);
      s1 = WMMA_F16(qa1, ke, s1);
    }

#pragma unroll
    for (int r = 0; r < 8; ++r) {
      float a0 = s0[r] * SCALE_;
      float a1 = s1[r] * SCALE_;
      float mx = grp16_max(fmaxf(a0, a1));
      float mnew  = fmaxf(mrun[r], mx);
      float alpha = __expf(mrun[r] - mnew);
      float p0 = __expf(a0 - mnew);
      float p1 = __expf(a1 - mnew);
      lrun[r] = lrun[r] * alpha + grp16_sum(p0 + p1);
      mrun[r] = mnew;
      o0[r] *= alpha; o1[r] *= alpha; o2[r] *= alpha; o3[r] *= alpha;
      const int row = r + hi8;
      pl[row * 32 + lr]      = (_Float16)p0;
      pl[row * 32 + 16 + lr] = (_Float16)p1;
    }
    __syncthreads();                    // P stores -> A-fragment loads

    v16h pa  = load_frag(pl, lr, 32, 0, lane);
    v16h vb0 = load_frag(vb + (size_t)0  * T_, lr, T_, kt, lane);
    v16h vb1 = load_frag(vb + (size_t)16 * T_, lr, T_, kt, lane);
    v16h vb2 = load_frag(vb + (size_t)32 * T_, lr, T_, kt, lane);
    v16h vb3 = load_frag(vb + (size_t)48 * T_, lr, T_, kt, lane);
    o0 = WMMA_F16(pa, vb0, o0);
    o1 = WMMA_F16(pa, vb1, o1);
    o2 = WMMA_F16(pa, vb2, o2);
    o3 = WMMA_F16(pa, vb3, o3);
    __syncthreads();                    // A-fragment loads -> next-iter stores
  }

#pragma unroll
  for (int r = 0; r < 8; ++r) {
    float inv = 1.0f / lrun[r];
    const int t = mt * 16 + r + hi8;
    const size_t ob = ((size_t)bb * T_ + t) * D_ + (size_t)h * HD_;
    ah[ob + 0  + lr] = (_Float16)(o0[r] * inv);
    ah[ob + 16 + lr] = (_Float16)(o1[r] * inv);
    ah[ob + 32 + lr] = (_Float16)(o2[r] * inv);
    ah[ob + 48 + lr] = (_Float16)(o3[r] * inv);
  }
}

// ---------------------------------------------------------------------------
// Stage 3: output projection out = attn @ Wo^T, same block tiling, fp32 store.
// ---------------------------------------------------------------------------
__global__ __launch_bounds__(256) void out_proj_wmma(
    const _Float16* __restrict__ ah,   // (B*T, D)
    const _Float16* __restrict__ woh,  // (D, D)
    float* __restrict__ out) {         // (B*T, D)
  __shared__ _Float16 As[2][BM * BK];
  __shared__ _Float16 Bs[2][BN * BK];
  const int tid  = threadIdx.x;
  const int lane = tid & 31;
  const int wave = tid >> 5;
  const int lr   = lane & 15;
  const int hi8  = (lane & 16) ? 8 : 0;
  const int m0   = blockIdx.x * BM;
  const int n0   = blockIdx.y * BN;
  const int wm   = (wave & 1) * 32;
  const int wn   = (wave >> 1) * 32;

  const _Float16* wbase = woh + (size_t)n0 * D_;
  const int cr = tid >> 2;
  const int cc = (tid & 3) * 8;

  v8f a00 = {}, a01 = {}, a10 = {}, a11 = {};

  auto fill = [&](int buf, int k0) {
    ASYNC_CP16(ah + (size_t)(m0 + cr) * D_ + k0 + cc, &As[buf][cr * BK + cc]);
    ASYNC_CP16(wbase + (size_t)cr * D_ + k0 + cc, &Bs[buf][cr * BK + cc]);
    ASYNC_CP16(wbase + (size_t)(64 + cr) * D_ + k0 + cc,
               &Bs[buf][(64 + cr) * BK + cc]);
  };

  fill(0, 0);
  ASYNC_WAIT();
  __syncthreads();

  for (int k0 = 0; k0 < D_; k0 += BK) {
    const int buf = (k0 >> 5) & 1;
    if (k0 + BK < D_) fill(buf ^ 1, k0 + BK);
    v16h fa0 = load_frag(As[buf], wm + lr,      BK, 0, lane);
    v16h fa1 = load_frag(As[buf], wm + 16 + lr, BK, 0, lane);
    v16h fb0 = load_frag(Bs[buf], wn + lr,      BK, 0, lane);
    v16h fb1 = load_frag(Bs[buf], wn + 16 + lr, BK, 0, lane);
    a00 = WMMA_F16(fa0, fb0, a00);
    a01 = WMMA_F16(fa0, fb1, a01);
    a10 = WMMA_F16(fa1, fb0, a10);
    a11 = WMMA_F16(fa1, fb1, a11);
    ASYNC_WAIT();
    __syncthreads();
  }

#pragma unroll
  for (int r = 0; r < 8; ++r) {
    const int m = m0 + wm + r + hi8;
    out[(size_t)m * D_ + n0 + wn + lr]             = a00[r];
    out[(size_t)m * D_ + n0 + wn + 16 + lr]        = a01[r];
    out[(size_t)(m + 16) * D_ + n0 + wn + lr]      = a10[r];
    out[(size_t)(m + 16) * D_ + n0 + wn + 16 + lr] = a11[r];
  }
}

// ---------------------------------------------------------------------------
extern "C" void kernel_launch(void* const* d_in, const int* in_sizes, int n_in,
                              void* d_out, int out_size, void* d_ws, size_t ws_size,
                              hipStream_t stream) {
  const float* x    = (const float*)d_in[0];
  const float* cosp = (const float*)d_in[1];
  const float* sinp = (const float*)d_in[2];
  const float* Wq   = (const float*)d_in[3];
  const float* Wkv  = (const float*)d_in[4];
  const float* Wo   = (const float*)d_in[5];
  float* out = (float*)d_out;

  _Float16* ws   = (_Float16*)d_ws;
  _Float16* xh   = ws;                                       // B*T*D
  _Float16* wqh  = xh   + (size_t)B_ * T_ * D_;              // D*D
  _Float16* wkvh = wqh  + (size_t)D_ * D_;                   // KVD*D
  _Float16* woh  = wkvh + (size_t)KVD_ * D_;                 // D*D
  _Float16* qh   = woh  + (size_t)D_ * D_;                   // B*QH*T*HD
  _Float16* kh   = qh   + (size_t)B_ * QH_ * T_ * HD_;       // B*KVH*T*HD
  _Float16* vt   = kh   + (size_t)B_ * KVH_ * T_ * HD_;      // B*KVH*HD*T
  _Float16* ah   = vt   + (size_t)B_ * KVH_ * HD_ * T_;      // B*T*D

  const int nx = B_ * T_ * D_;
  f32_to_f16_kernel<<<(nx + 255) / 256, 256, 0, stream>>>(x, xh, nx);
  f32_to_f16_kernel<<<(D_ * D_ + 255) / 256, 256, 0, stream>>>(Wq, wqh, D_ * D_);
  f32_to_f16_kernel<<<(KVD_ * D_ + 255) / 256, 256, 0, stream>>>(Wkv, wkvh, KVD_ * D_);
  f32_to_f16_kernel<<<(D_ * D_ + 255) / 256, 256, 0, stream>>>(Wo, woh, D_ * D_);

  dim3 g1(B_ * T_ / BM, (D_ + KVD_) / BN);        // (64, 12)
  qkv_rope_wmma<<<g1, 256, 0, stream>>>(xh, wqh, wkvh, cosp, sinp, qh, kh, vt);

  dim3 g2(T_ / 16, QH_, B_);                      // (128, 16, 2)
  attn_wmma<<<g2, 32, 0, stream>>>(qh, kh, vt, ah);

  dim3 g3(B_ * T_ / BM, D_ / BN);                 // (64, 8)
  out_proj_wmma<<<g3, 256, 0, stream>>>(ah, woh, out);
}
